// SelfAttention_89429809037579
// MI455X (gfx1250) — matrix operand order repaired and checked by reading, not compiled, not run on hardware
//
#include <hip/hip_runtime.h>

// ---------------------------------------------------------------------------
// MI455X (gfx1250) fused self-attention, f16 WMMA with f32 accumulation.
//   prep:   x -> f16, W* -> transposed f16 (column-major so B-frags load flat)
//   gemm:   q,k = xh@W (f16 out; Q pre-scaled by 0.125*log2e so the softmax
//           runs in the exp2 domain), v stored transposed [B,H,D,S]
//           4x4 WMMA blocking: 16 v_wmma per 16 b128 loads per k-step
//   attn:   flash online softmax (base-2), 32 query rows/wave, additive mask
//           bias, row-sums via ones-WMMA, probs re-laid out through LDS
//   gemm:   out = ctx@Wo + bo (f32 out)
// Inter-kernel working set (~48MB f16) stays resident in the 192MB L2.
// ---------------------------------------------------------------------------

typedef _Float16 half8 __attribute__((ext_vector_type(8)));
typedef _Float16 v16h  __attribute__((ext_vector_type(16)));
typedef float    v8f   __attribute__((ext_vector_type(8)));

#define AB 2
#define AS 2048
#define AE 1024
#define AH 16
#define AD 64
#define NEG_BIG (-3.0e38f)
// 1/sqrt(D) * log2(e) : folds both the attention scale and the exp->exp2
// conversion into the Q projection epilogue.
#define Q_ALPHA (0.125f * 1.44269504088896340736f)

__device__ __forceinline__ int laneid() { return (int)(threadIdx.x & 31u); }

// A-fragment 16x32 f16 (row-major source, leading dim ld):
//   lanes 0-15 : row = lane,    K = k0+{0..7, 16..23}
//   lanes 16-31: row = lane-16, K = k0+{8..15, 24..31}
__device__ __forceinline__ v16h load_a(const _Float16* __restrict__ p, int ld, int r0, int k0) {
  int lane = laneid(); int hi = lane >> 4; int r = lane & 15;
  const _Float16* base = p + (size_t)(r0 + r) * (size_t)ld + (size_t)(k0 + hi * 8);
  half8 lo = *(const half8*)(base);
  half8 hh = *(const half8*)(base + 16);
  v16h out;
#pragma unroll
  for (int i = 0; i < 8; ++i) { out[i] = lo[i]; out[i + 8] = hh[i]; }
  return out;
}

// B-fragment 32x16 f16 from BT stored as [N][K] row-major:
//   lanes 0-15 : col = lane,    K = k0+{0..7, 16..23}
//   lanes 16-31: col = lane-16, K = k0+{8..15, 24..31}
__device__ __forceinline__ v16h load_b(const _Float16* __restrict__ p, int ld, int n0, int k0) {
  int lane = laneid(); int hi = lane >> 4; int n = lane & 15;
  const _Float16* base = p + (size_t)(n0 + n) * (size_t)ld + (size_t)(k0 + hi * 8);
  half8 lo = *(const half8*)(base);
  half8 hh = *(const half8*)(base + 16);
  v16h out;
#pragma unroll
  for (int i = 0; i < 8; ++i) { out[i] = lo[i]; out[i + 8] = hh[i]; }
  return out;
}

__device__ __forceinline__ v8f wmma_f16(v16h a, v16h b, v8f c) {
  return __builtin_amdgcn_wmma_f32_16x16x32_f16(false, a, false, b, (short)0, c, false, false);
}

// max-reduction across the 16-lane half (rows 0-7 in lanes 0-15, 8-15 in 16-31)
__device__ __forceinline__ float redmax16(float v) {
#pragma unroll
  for (int m = 1; m <= 8; m <<= 1) v = fmaxf(v, __shfl_xor(v, m, 32));
  return v;
}

// ---------------------------------------------------------------------------
__global__ void f32_to_f16_kernel(const float* __restrict__ in, _Float16* __restrict__ out, int n) {
  int i = blockIdx.x * blockDim.x + threadIdx.x;
  if (i < n) out[i] = (_Float16)in[i];
}

// WT[n*E + k] = W[k*E + n]  (f32 -> f16)
__global__ void transpose_w_kernel(const float* __restrict__ W, _Float16* __restrict__ WT) {
  int i = blockIdx.x * blockDim.x + threadIdx.x;
  int n = i >> 10, k = i & (AE - 1);
  WT[i] = (_Float16)W[(size_t)k * AE + n];
}

// ---------------------------------------------------------------------------
// C = alpha*(A[M,K] @ BT[N,K]^T + bias).  One wave -> 64x64 tile (4x4 WMMA
// blocking), 4 waves/block arranged 2x2 over a 128x128 block tile.
// MODE 0: f16 row-major out.  MODE 1: f16 transposed out [b][col][s] (for V).
// MODE 2: f32 row-major out.
template <int MODE>
__global__ __launch_bounds__(128) void gemm_kernel(const _Float16* __restrict__ A,
                                                   const _Float16* __restrict__ BT,
                                                   const float* __restrict__ bias,
                                                   void* __restrict__ out,
                                                   int N, int K, int batch_rows,
                                                   float alpha) {
  int wid = threadIdx.x >> 5;
  int m0 = blockIdx.y * 128 + (wid >> 1) * 64;
  int n0 = blockIdx.x * 128 + (wid & 1) * 64;
  v8f acc[4][4] = {};
  for (int k0 = 0; k0 < K; k0 += 32) {
    v16h a[4], bf[4];
#pragma unroll
    for (int mi = 0; mi < 4; ++mi) a[mi] = load_a(A, K, m0 + mi * 16, k0);
#pragma unroll
    for (int ni = 0; ni < 4; ++ni) bf[ni] = load_b(BT, K, n0 + ni * 16, k0);
#pragma unroll
    for (int mi = 0; mi < 4; ++mi)
#pragma unroll
      for (int ni = 0; ni < 4; ++ni)
        acc[mi][ni] = wmma_f16(a[mi], bf[ni], acc[mi][ni]);
  }
  int lane = laneid(); int hi = lane >> 4; int n = lane & 15;
#pragma unroll
  for (int mi = 0; mi < 4; ++mi)
#pragma unroll
    for (int ni = 0; ni < 4; ++ni) {
      int col = n0 + ni * 16 + n;
      float bn = bias[col];
      int row0 = m0 + mi * 16 + hi * 8;
      if (MODE == 0) {
        _Float16* o = (_Float16*)out;
#pragma unroll
        for (int i = 0; i < 8; ++i)
          o[(size_t)(row0 + i) * N + col] = (_Float16)((acc[mi][ni][i] + bn) * alpha);
      } else if (MODE == 1) {
        int bb = row0 / batch_rows, s = row0 % batch_rows;
        half8 hv;
#pragma unroll
        for (int i = 0; i < 8; ++i) hv[i] = (_Float16)((acc[mi][ni][i] + bn) * alpha);
        *(half8*)((_Float16*)out + ((size_t)bb * N + col) * (size_t)batch_rows + s) = hv;
      } else {
        float* o = (float*)out;
#pragma unroll
        for (int i = 0; i < 8; ++i)
          o[(size_t)(row0 + i) * N + col] = (acc[mi][ni][i] + bn) * alpha;
      }
    }
}

// ---------------------------------------------------------------------------
// Flash attention: one wave per (b, h, 32-query-row tile); two 16-row WMMA
// tiles share the K/V fragments. Scores arrive already in the exp2 domain
// (Q pre-scaled by 0.125*log2e). Masking is an additive -3e38 bias; exp2
// underflow makes masked probabilities exactly 0 once a row has any real key,
// and the online correction factor wipes any all-masked prefix, so this
// matches the reference's post-softmax mask renormalization.
// Row sums come from a ones-matrix WMMA (P @ 1 lands per-row sums exactly in
// the C-layout slots of rsum).
__global__ __launch_bounds__(128) void attn_kernel(const _Float16* __restrict__ qh,
                                                   const _Float16* __restrict__ kh,
                                                   const _Float16* __restrict__ vT,
                                                   const int* __restrict__ mask,
                                                   _Float16* __restrict__ ctxh) {
  __shared__ _Float16 plds[4][32 * 32];  // per-wave 32x32 prob tile
  int wid = threadIdx.x >> 5;
  int lane = laneid(); int hi = lane >> 4; int ln = lane & 15;
  int b = blockIdx.z, h = blockIdx.y;
  int m0 = blockIdx.x * 128 + wid * 32;

  const _Float16* qb = qh + (size_t)b * AS * AE + h * AD;
  const _Float16* kb = kh + (size_t)b * AS * AE + h * AD;
  const _Float16* vb = vT + ((size_t)b * AH + h) * (size_t)AD * AS;
  const int* mb = mask + (size_t)b * AS;

  v16h qa[2][2];
#pragma unroll
  for (int t = 0; t < 2; ++t) {
    qa[t][0] = load_a(qb, AE, m0 + t * 16, 0);
    qa[t][1] = load_a(qb, AE, m0 + t * 16, 32);
  }
  v8f cacc[2][4] = {};
  float rmax[2][8], rsum[2][8];
#pragma unroll
  for (int t = 0; t < 2; ++t)
#pragma unroll
    for (int i = 0; i < 8; ++i) { rmax[t][i] = NEG_BIG; rsum[t][i] = 0.f; }

  v16h onesb;
#pragma unroll
  for (int i = 0; i < 16; ++i) onesb[i] = (_Float16)1.0f;

  const v8f vzero = {};

  for (int n0 = 0; n0 < AS; n0 += 32) {
    v16h kb0a = load_b(kb, AE, n0, 0);
    v16h kb0b = load_b(kb, AE, n0, 32);
    v16h kb1a = load_b(kb, AE, n0 + 16, 0);
    v16h kb1b = load_b(kb, AE, n0 + 16, 32);
    float add0 = mb[n0 + ln] ? 0.f : NEG_BIG;       // additive mask bias
    float add1 = mb[n0 + 16 + ln] ? 0.f : NEG_BIG;
    _Float16* pl = plds[wid];
    v16h pa[2];

#pragma unroll
    for (int t = 0; t < 2; ++t) {
      v8f s0 = wmma_f16(qa[t][0], kb0a, vzero);
      s0 = wmma_f16(qa[t][1], kb0b, s0);
      v8f s1 = wmma_f16(qa[t][0], kb1a, vzero);
      s1 = wmma_f16(qa[t][1], kb1b, s1);
      float corr[8];
#pragma unroll
      for (int i = 0; i < 8; ++i) {
        float f0 = s0[i] + add0;
        float f1 = s1[i] + add1;
        float cm = redmax16(fmaxf(f0, f1));
        float nmax = fmaxf(rmax[t][i], cm);
        corr[i] = exp2f(rmax[t][i] - nmax);
        rmax[t][i] = nmax;
        float p0 = exp2f(f0 - nmax);   // exact 0 for masked keys (underflow)
        float p1 = exp2f(f1 - nmax);
        int row = t * 16 + hi * 8 + i;
        pl[row * 32 + ln] = (_Float16)p0;
        pl[row * 32 + 16 + ln] = (_Float16)p1;
      }
#pragma unroll
      for (int j = 0; j < 4; ++j)
#pragma unroll
        for (int i = 0; i < 8; ++i) cacc[t][j][i] *= corr[i];
#pragma unroll
      for (int i = 0; i < 8; ++i) rsum[t][i] *= corr[i];

      // C-layout -> A-fragment re-layout through LDS (in-order within a wave)
      const _Float16* pbase = &pl[(t * 16 + ln) * 32 + hi * 8];
      half8 plo = *(const half8*)(pbase);
      half8 phi = *(const half8*)(pbase + 16);
#pragma unroll
      for (int i = 0; i < 8; ++i) { pa[t][i] = plo[i]; pa[t][i + 8] = phi[i]; }

      // row sums of this 32-key chunk via ones-matrix WMMA
      v8f csum = wmma_f16(pa[t], onesb, vzero);
#pragma unroll
      for (int i = 0; i < 8; ++i) rsum[t][i] += csum[i];
    }

#pragma unroll
    for (int j = 0; j < 4; ++j) {
      v16h vbf = load_b(vb, AS, j * 16, n0);
      cacc[0][j] = wmma_f16(pa[0], vbf, cacc[0][j]);
      cacc[1][j] = wmma_f16(pa[1], vbf, cacc[1][j]);
    }
  }

#pragma unroll
  for (int t = 0; t < 2; ++t) {
    float inv[8];
#pragma unroll
    for (int i = 0; i < 8; ++i) inv[i] = rsum[t][i] > 0.f ? 1.0f / rsum[t][i] : 0.f;
    _Float16* cb = ctxh + ((size_t)b * AS + m0 + t * 16) * AE + h * AD;
#pragma unroll
    for (int j = 0; j < 4; ++j)
#pragma unroll
      for (int i = 0; i < 8; ++i)
        cb[(size_t)(hi * 8 + i) * AE + j * 16 + ln] = (_Float16)(cacc[t][j][i] * inv[i]);
  }
}

// ---------------------------------------------------------------------------
extern "C" void kernel_launch(void* const* d_in, const int* in_sizes, int n_in,
                              void* d_out, int out_size, void* d_ws, size_t ws_size,
                              hipStream_t stream) {
  const float* x  = (const float*)d_in[0];
  const int* mask = (const int*)d_in[1];
  const float* Wq = (const float*)d_in[2];
  const float* bq = (const float*)d_in[3];
  const float* Wk = (const float*)d_in[4];
  const float* bk = (const float*)d_in[5];
  const float* Wv = (const float*)d_in[6];
  const float* bv = (const float*)d_in[7];
  const float* Wo = (const float*)d_in[8];
  const float* bo = (const float*)d_in[9];
  float* out = (float*)d_out;

  const int B = AB, S = AS, E = AE, H = AH;
  const int M = B * S;  // 4096

  _Float16* ws   = (_Float16*)d_ws;
  _Float16* xh   = ws;                       // M*E
  _Float16* qh   = xh   + (size_t)M * E;     // M*E (pre-scaled by Q_ALPHA)
  _Float16* khp  = qh   + (size_t)M * E;     // M*E
  _Float16* vT   = khp  + (size_t)M * E;     // M*E, layout [B][H][D][S]
  _Float16* ctxh = vT   + (size_t)M * E;     // M*E
  _Float16* WqT  = ctxh + (size_t)M * E;     // E*E
  _Float16* WkT  = WqT  + (size_t)E * E;
  _Float16* WvT  = WkT  + (size_t)E * E;
  _Float16* WoT  = WvT  + (size_t)E * E;

  f32_to_f16_kernel<<<(M * E) / 256, 256, 0, stream>>>(x, xh, M * E);
  transpose_w_kernel<<<(E * E) / 256, 256, 0, stream>>>(Wq, WqT);
  transpose_w_kernel<<<(E * E) / 256, 256, 0, stream>>>(Wk, WkT);
  transpose_w_kernel<<<(E * E) / 256, 256, 0, stream>>>(Wv, WvT);
  transpose_w_kernel<<<(E * E) / 256, 256, 0, stream>>>(Wo, WoT);

  dim3 gg(E / 128, M / 128);
  gemm_kernel<0><<<gg, 128, 0, stream>>>(xh, WqT, bq, qh,  E, E, S, Q_ALPHA);
  gemm_kernel<0><<<gg, 128, 0, stream>>>(xh, WkT, bk, khp, E, E, S, 1.0f);
  gemm_kernel<1><<<gg, 128, 0, stream>>>(xh, WvT, bv, vT,  E, E, S, 1.0f);

  attn_kernel<<<dim3(S / 128, H, B), 128, 0, stream>>>(qh, khp, vT, mask, ctxh);

  gemm_kernel<2><<<gg, 128, 0, stream>>>(ctxh, WoT, bo, out, E, E, S, 1.0f);
}
